// HybridGnnTorso_74036646248565
// MI455X (gfx1250) — matrix operand order repaired and checked by reading, not compiled, not run on hardware
//
#include <hip/hip_runtime.h>

// ---------------------------------------------------------------------------
// HybridGnnTorso for MI455X (gfx1250, wave32, WMMA).
// All GEMMs use v_wmma_f32_16x16x32_f16 with fp16 activations/weights and
// fp32 accumulation. Weights are pre-packed (kt-major) into the per-lane WMMA
// B-fragment layout; the GEMM holds 8 n-tile accumulators so each A fragment
// is loaded once and feeds 8 WMMAs (k-outer / n-inner).
// ---------------------------------------------------------------------------

typedef _Float16 v8h  __attribute__((ext_vector_type(8)));
typedef _Float16 v16h __attribute__((ext_vector_type(16)));
typedef float    v8f  __attribute__((ext_vector_type(8)));

#define BN 64
#define TN 8
#define SN 16
#define CN 128
#define LN_LAYERS 3
#define NNODE (BN * SN * SN * SN)   // 262144
#define ACT_ROWS (BN * 7)           // 448

// ---------------------------------------------------------------------------
// Pack a K x 128 (row-major, ld = 128) f32 weight matrix into the WMMA
// B-fragment layout for V_WMMA_F32_16X16X32_F16, kt-major:
// Packed element index: ((kt*8 + nt)*32 + lane)*16 + v*2 + p
//   k = kt*32 + (lane>>4)*16 + 2v + p ,  n = nt*16 + (lane&15)
// ---------------------------------------------------------------------------
__global__ void pack_kernel(const float* __restrict__ W, _Float16* __restrict__ P,
                            int Ktiles, int ldw)
{
    const int t = blockIdx.x * blockDim.x + threadIdx.x;
    const int total = Ktiles * 8 * 32 * 8;   // dword-pairs
    if (t >= total) return;
    const int v    = t & 7;
    const int lane = (t >> 3) & 31;
    const int rest = t >> 8;                 // kt*8 + nt
    const int nt   = rest & 7;
    const int kt   = rest >> 3;
    const int k    = kt * 32 + (lane >> 4) * 16 + 2 * v;
    const int n    = nt * 16 + (lane & 15);
    P[(size_t)t * 2 + 0] = (_Float16)W[(size_t)k * ldw + n];
    P[(size_t)t * 2 + 1] = (_Float16)W[(size_t)(k + 1) * ldw + n];
}

// ---------------------------------------------------------------------------
// Generic WMMA GEMM:  out[M x 128] = A[M x K] (f16, row-major, stride ldA)
//                                  @ Bp (packed f16, kt-major)  (+ bias, relu)
// One wave computes a full 16x128 output stripe: 8 resident accumulators,
// k-outer / n-inner so each A fragment is loaded once per k-step.
// ---------------------------------------------------------------------------
__global__ void __launch_bounds__(128)
wmma_gemm_kernel(const _Float16* __restrict__ A,
                 const _Float16* __restrict__ Bp,
                 const float* __restrict__ bias,
                 float* __restrict__ outF,
                 _Float16* __restrict__ outH,
                 int Mtiles, int K, int ldA, int relu)
{
    const int lane    = threadIdx.x & 31;
    const int wave    = threadIdx.x >> 5;
    const int rowTile = blockIdx.x * 4 + wave;
    if (rowTile >= Mtiles) return;           // wave-uniform; EXEC stays all-1s

    const int m      = lane & 15;
    const int half   = lane >> 4;
    const int Ktiles = K >> 5;
    const _Float16* Arow = A + (size_t)(rowTile * 16 + m) * ldA;

    v8f acc[8];
    const v8f zero = {};
    #pragma unroll
    for (int nt = 0; nt < 8; ++nt) acc[nt] = zero;

    const _Float16* bp = Bp + (size_t)lane * 16;
    for (int kt = 0; kt < Ktiles; ++kt) {
        // A fragment (16x32 f16): lanes 0-15 -> K 0..15,
        //                         lanes 16-31 -> K 16..31
        const int col = (kt << 5) + half * 16;
        v8h alo = *(const v8h*)(Arow + col);
        v8h ahi = *(const v8h*)(Arow + col + 8);
        v16h a = __builtin_shufflevector(alo, ahi,
                   0,1,2,3,4,5,6,7,8,9,10,11,12,13,14,15);
        #pragma unroll
        for (int nt = 0; nt < 8; ++nt) {
            // B fragment: 32 contiguous bytes per lane; nt stride = 512 halfs
            v8h blo = *(const v8h*)(bp + nt * 512);
            v8h bhi = *(const v8h*)(bp + nt * 512 + 8);
            v16h b = __builtin_shufflevector(blo, bhi,
                       0,1,2,3,4,5,6,7,8,9,10,11,12,13,14,15);
            acc[nt] = __builtin_amdgcn_wmma_f32_16x16x32_f16(
                          false, a, false, b, (short)0, acc[nt], false, false);
        }
        bp += 8 * 512;                       // next kt block (4096 halfs)
    }

    // C/D layout: lane col n = lane&15, rows M = half*8 + v
    #pragma unroll
    for (int nt = 0; nt < 8; ++nt) {
        const int ncol = nt * 16 + m;
        const float bv = bias ? bias[ncol] : 0.0f;
        #pragma unroll
        for (int v = 0; v < 8; ++v) {
            float val = acc[nt][v] + bv;
            if (relu) val = fmaxf(val, 0.0f);
            const size_t idx = (size_t)(rowTile * 16 + half * 8 + v) * CN + ncol;
            if (outF) outF[idx] = val;
            else      outH[idx] = (_Float16)val;
        }
    }
}

// ---------------------------------------------------------------------------
// Featurize: h0 = (feat @ W_feat + b_feat) * mask  (f16 out)
// ---------------------------------------------------------------------------
__global__ void __launch_bounds__(128)
featurize_kernel(const float* __restrict__ xx, const float* __restrict__ ss,
                 const float* __restrict__ Wf, const float* __restrict__ bf,
                 _Float16* __restrict__ h)
{
    const int c    = threadIdx.x;
    const int node = blockIdx.x;
    const int b    = node >> 12;
    const int cell = node & 4095;
    const int i = cell >> 8, j = (cell >> 4) & 15, k = cell & 15;
    const float val  = xx[(size_t)b * (TN * 4096) + cell];
    const float mask = (val != 0.0f) ? 1.0f : 0.0f;
    const float inv15 = 1.0f / 15.0f;
    float hv = ((float)i * inv15) * Wf[c]
             + ((float)j * inv15) * Wf[CN + c]
             + ((float)k * inv15) * Wf[2 * CN + c]
             + (val * 0.5f)       * Wf[3 * CN + c]
             + (ss[b] * (1.0f / 16.0f)) * Wf[4 * CN + c]
             + bf[c];
    h[(size_t)node * CN + c] = (_Float16)(hv * mask);
}

// ---------------------------------------------------------------------------
// Slab counts of the mask: cnt[b*48 + {i | 16+j | 32+k}]
// ---------------------------------------------------------------------------
__global__ void __launch_bounds__(256)
cnt_kernel(const float* __restrict__ xx, float* __restrict__ cnt)
{
    __shared__ float lc[48];
    const int b = blockIdx.x;
    if (threadIdx.x < 48) lc[threadIdx.x] = 0.0f;
    __syncthreads();
    for (int cell = threadIdx.x; cell < 4096; cell += blockDim.x) {
        const float v = xx[(size_t)b * (TN * 4096) + cell];
        if (v != 0.0f) {
            const int i = cell >> 8, j = (cell >> 4) & 15, k = cell & 15;
            atomicAdd(&lc[i], 1.0f);
            atomicAdd(&lc[16 + j], 1.0f);
            atomicAdd(&lc[32 + k], 1.0f);
        }
    }
    __syncthreads();
    if (threadIdx.x < 48) cnt[b * 48 + threadIdx.x] = lc[threadIdx.x];
}

// ---------------------------------------------------------------------------
// Slab sums of msg_r: sums[(b*3+rel)*16 + idx][c] = sum over the 256 cells
// blockIdx.x = b*48 + rel*16 + idx  (== (b*3+rel)*16+idx linearized)
// ---------------------------------------------------------------------------
__global__ void __launch_bounds__(128)
slabsum_kernel(const _Float16* __restrict__ m0, const _Float16* __restrict__ m1,
               const _Float16* __restrict__ m2, float* __restrict__ sums)
{
    const int c   = threadIdx.x;
    const int gb  = blockIdx.x;
    const int b   = gb / 48;
    const int rem = gb % 48;
    const int rel = rem >> 4;
    const int idx = rem & 15;
    const _Float16* msg = (rel == 0) ? m0 : ((rel == 1) ? m1 : m2);
    const size_t base = (size_t)b * 4096;
    float acc = 0.0f;
    for (int t = 0; t < 256; ++t) {
        size_t node;
        if (rel == 0)      node = base + (size_t)idx * 256 + t;
        else if (rel == 1) node = base + (size_t)(t >> 4) * 256 + idx * 16 + (t & 15);
        else               node = base + (size_t)(t >> 4) * 256 + (t & 15) * 16 + idx;
        acc += (float)msg[node * CN + c];
    }
    sums[(size_t)gb * CN + c] = acc;
}

// ---------------------------------------------------------------------------
// Combine: out = yroot + b_conv + sum_r where(deg>0, (slabsum-msg)/max(deg,1),0)
// then h = layernorm(relu(out)) * mask   (f16 out)
// ---------------------------------------------------------------------------
__global__ void __launch_bounds__(128)
combine_kernel(const float* __restrict__ yroot,
               const _Float16* __restrict__ msg0, const _Float16* __restrict__ msg1,
               const _Float16* __restrict__ msg2,
               const float* __restrict__ sums, const float* __restrict__ cnt,
               const float* __restrict__ xx, const float* __restrict__ bconv,
               const float* __restrict__ lng, const float* __restrict__ lnb,
               _Float16* __restrict__ h)
{
    const int c    = threadIdx.x;
    const int node = blockIdx.x;
    const int b    = node >> 12;
    const int cell = node & 4095;
    const int i = cell >> 8, j = (cell >> 4) & 15, k = cell & 15;
    const float val  = xx[(size_t)b * (TN * 4096) + cell];
    const float mask = (val != 0.0f) ? 1.0f : 0.0f;
    const size_t nc  = (size_t)node * CN + c;

    float o = yroot[nc] + bconv[c];
    {   // rel 0: neighbors share i
        const float deg = cnt[b * 48 + i] - mask;
        const float agg = sums[((size_t)(b * 3 + 0) * 16 + i) * CN + c] - (float)msg0[nc];
        o += (deg > 0.0f) ? agg / fmaxf(deg, 1.0f) : 0.0f;
    }
    {   // rel 1: neighbors share j
        const float deg = cnt[b * 48 + 16 + j] - mask;
        const float agg = sums[((size_t)(b * 3 + 1) * 16 + j) * CN + c] - (float)msg1[nc];
        o += (deg > 0.0f) ? agg / fmaxf(deg, 1.0f) : 0.0f;
    }
    {   // rel 2: neighbors share k
        const float deg = cnt[b * 48 + 32 + k] - mask;
        const float agg = sums[((size_t)(b * 3 + 2) * 16 + k) * CN + c] - (float)msg2[nc];
        o += (deg > 0.0f) ? agg / fmaxf(deg, 1.0f) : 0.0f;
    }
    o = fmaxf(o, 0.0f);

    // LayerNorm over C=128 (4 waves): wave shfl reduction + LDS combine
    __shared__ float red0[4];
    __shared__ float red1[4];
    float s = o;
    #pragma unroll
    for (int off = 16; off; off >>= 1) s += __shfl_xor(s, off);
    if ((threadIdx.x & 31) == 0) red0[threadIdx.x >> 5] = s;
    __syncthreads();
    const float mean = (red0[0] + red0[1] + red0[2] + red0[3]) * (1.0f / 128.0f);
    const float d = o - mean;
    float q = d * d;
    #pragma unroll
    for (int off = 16; off; off >>= 1) q += __shfl_xor(q, off);
    if ((threadIdx.x & 31) == 0) red1[threadIdx.x >> 5] = q;
    __syncthreads();
    const float var = (red1[0] + red1[1] + red1[2] + red1[3]) * (1.0f / 128.0f);
    const float y = d * rsqrtf(var + 1e-5f) * lng[c] + lnb[c];
    h[nc] = (_Float16)(y * mask);
}

// ---------------------------------------------------------------------------
// Pooling means: rows 0..255 = mean over i, 256..511 = mean over j,
// 512..767 = mean over k. Output row stride 770 per batch.
// ---------------------------------------------------------------------------
__global__ void __launch_bounds__(128)
pool_kernel(const _Float16* __restrict__ h, float* __restrict__ out)
{
    const int c   = threadIdx.x;
    const int gb  = blockIdx.x;          // b*768 + row
    const int b   = gb / 768;
    const int row = gb % 768;
    const int sec = row >> 8;
    const int rc  = row & 255;
    const int p = rc >> 4, q = rc & 15;
    const size_t base = (size_t)b * 4096;
    float acc = 0.0f;
    #pragma unroll
    for (int t = 0; t < 16; ++t) {
        size_t node;
        if (sec == 0)      node = base + (size_t)t * 256 + p * 16 + q;   // mean over i
        else if (sec == 1) node = base + (size_t)p * 256 + t * 16 + q;   // mean over j
        else               node = base + (size_t)p * 256 + q * 16 + t;   // mean over k
        acc += (float)h[node * CN + c];
    }
    out[((size_t)b * 770 + row) * CN + c] = acc * (1.0f / 16.0f);
}

// ---------------------------------------------------------------------------
// Action path prep: Aact[448 x 4096] f16 = xx[:, 1:8, :]
// ---------------------------------------------------------------------------
__global__ void __launch_bounds__(256)
actprep_kernel(const float* __restrict__ xx, _Float16* __restrict__ Aact)
{
    const size_t n = (size_t)blockIdx.x * blockDim.x + threadIdx.x;
    if (n >= (size_t)ACT_ROWS * 4096) return;
    const int rowi = (int)(n >> 12);
    const int cell = (int)(n & 4095);
    const int b = rowi / 7;
    const int t = rowi % 7;
    Aact[n] = (_Float16)xx[((size_t)b * TN + 1 + t) * 4096 + cell];
}

// ---------------------------------------------------------------------------
// Final rows: 768 = mean over t of act MLP output, 769 = relu(ss*W_s + b_s)
// ---------------------------------------------------------------------------
__global__ void __launch_bounds__(128)
actfinal_kernel(const float* __restrict__ actout, const float* __restrict__ ss,
                const float* __restrict__ Ws, const float* __restrict__ bs,
                float* __restrict__ out)
{
    const int c = threadIdx.x;
    const int b = blockIdx.x;
    float acc = 0.0f;
    #pragma unroll
    for (int t = 0; t < 7; ++t)
        acc += actout[((size_t)(b * 7 + t)) * CN + c];
    out[((size_t)b * 770 + 768) * CN + c] = acc * (1.0f / 7.0f);
    out[((size_t)b * 770 + 769) * CN + c] = fmaxf(ss[b] * Ws[c] + bs[c], 0.0f);
}

// ---------------------------------------------------------------------------
// Host driver
// ---------------------------------------------------------------------------
extern "C" void kernel_launch(void* const* d_in, const int* in_sizes, int n_in,
                              void* d_out, int out_size, void* d_ws, size_t ws_size,
                              hipStream_t stream)
{
    (void)in_sizes; (void)n_in; (void)out_size; (void)ws_size;

    const float* xx     = (const float*)d_in[0];
    const float* ss     = (const float*)d_in[1];
    const float* W_feat = (const float*)d_in[2];
    const float* b_feat = (const float*)d_in[3];
    const float* W_root = (const float*)d_in[4];
    const float* W_rel  = (const float*)d_in[5];
    const float* b_conv = (const float*)d_in[6];
    const float* ln_g   = (const float*)d_in[7];
    const float* ln_b   = (const float*)d_in[8];
    const float* W_a1   = (const float*)d_in[9];
    const float* b_a1   = (const float*)d_in[10];
    const float* W_a2   = (const float*)d_in[11];
    const float* b_a2   = (const float*)d_in[12];
    const float* W_s    = (const float*)d_in[13];
    const float* b_s    = (const float*)d_in[14];
    float* out = (float*)d_out;

    // workspace layout (all 256-byte aligned)
    char* ws = (char*)d_ws;
    size_t off = 0;
    auto alloc = [&](size_t bytes) -> void* {
        void* p = ws + off;
        off = (off + bytes + 255) & ~(size_t)255;
        return p;
    };
    _Float16* h      = (_Float16*)alloc((size_t)NNODE * CN * 2);        //  67 MB
    float*    yroot  = (float*)   alloc((size_t)NNODE * CN * 4);        // 134 MB
    _Float16* msg0   = (_Float16*)alloc((size_t)NNODE * CN * 2);
    _Float16* msg1   = (_Float16*)alloc((size_t)NNODE * CN * 2);
    _Float16* msg2   = (_Float16*)alloc((size_t)NNODE * CN * 2);
    float*    sums   = (float*)   alloc((size_t)BN * 3 * 16 * CN * 4);
    float*    cnt    = (float*)   alloc((size_t)BN * 48 * 4);
    _Float16* packL  = (_Float16*)alloc((size_t)12 * 16384 * 2);        // 12 x (128x128)
    _Float16* packA1 = (_Float16*)alloc((size_t)128 * 8 * 32 * 16 * 2); // 4096x128
    _Float16* packA2 = (_Float16*)alloc((size_t)16384 * 2);             // 128x128
    _Float16* Aact   = (_Float16*)alloc((size_t)ACT_ROWS * 4096 * 2);
    _Float16* hidden = (_Float16*)alloc((size_t)ACT_ROWS * CN * 2);
    float*    actout = (float*)   alloc((size_t)ACT_ROWS * CN * 4);

    // --- pack all weight matrices into WMMA B-fragment layout (kt-major) ---
    const int pack128 = (4 * 8 * 32 * 8 + 255) / 256;    // 128x128 matrices
    for (int l = 0; l < LN_LAYERS; ++l) {
        pack_kernel<<<pack128, 256, 0, stream>>>(
            W_root + (size_t)l * CN * CN, packL + (size_t)(l * 4) * 16384, 4, CN);
        for (int r = 0; r < 3; ++r)
            pack_kernel<<<pack128, 256, 0, stream>>>(
                W_rel + (size_t)(l * 3 + r) * CN * CN,
                packL + (size_t)(l * 4 + 1 + r) * 16384, 4, CN);
    }
    pack_kernel<<<(128 * 8 * 32 * 8 + 255) / 256, 256, 0, stream>>>(W_a1, packA1, 128, CN);
    pack_kernel<<<pack128, 256, 0, stream>>>(W_a2, packA2, 4, CN);

    // --- featurize + slab counts ---
    cnt_kernel<<<BN, 256, 0, stream>>>(xx, cnt);
    featurize_kernel<<<NNODE, 128, 0, stream>>>(xx, ss, W_feat, b_feat, h);

    // --- GNN layers ---
    const int Mtiles = NNODE / 16;          // 16384
    const int gemmGrid = Mtiles / 4;        // 4096 blocks x 4 waves
    for (int l = 0; l < LN_LAYERS; ++l) {
        wmma_gemm_kernel<<<gemmGrid, 128, 0, stream>>>(
            h, packL + (size_t)(l * 4 + 0) * 16384, nullptr, yroot, nullptr,
            Mtiles, CN, CN, 0);
        wmma_gemm_kernel<<<gemmGrid, 128, 0, stream>>>(
            h, packL + (size_t)(l * 4 + 1) * 16384, nullptr, nullptr, msg0,
            Mtiles, CN, CN, 0);
        wmma_gemm_kernel<<<gemmGrid, 128, 0, stream>>>(
            h, packL + (size_t)(l * 4 + 2) * 16384, nullptr, nullptr, msg1,
            Mtiles, CN, CN, 0);
        wmma_gemm_kernel<<<gemmGrid, 128, 0, stream>>>(
            h, packL + (size_t)(l * 4 + 3) * 16384, nullptr, nullptr, msg2,
            Mtiles, CN, CN, 0);
        slabsum_kernel<<<BN * 48, 128, 0, stream>>>(msg0, msg1, msg2, sums);
        combine_kernel<<<NNODE, 128, 0, stream>>>(
            yroot, msg0, msg1, msg2, sums, cnt, xx,
            b_conv + (size_t)l * CN, ln_g, ln_b, h);
    }

    // --- pooling means -> output rows 0..767 ---
    pool_kernel<<<BN * 768, 128, 0, stream>>>(h, out);

    // --- action MLP (WMMA) + final rows ---
    actprep_kernel<<<(ACT_ROWS * 4096 + 255) / 256, 256, 0, stream>>>(xx, Aact);
    wmma_gemm_kernel<<<7, 128, 0, stream>>>(   // 448x4096 @ 4096x128, relu, f16 out
        Aact, packA1, b_a1, nullptr, hidden, ACT_ROWS / 16, 4096, 4096, 1);
    wmma_gemm_kernel<<<7, 128, 0, stream>>>(   // 448x128 @ 128x128, f32 out
        hidden, packA2, b_a2, actout, nullptr, ACT_ROWS / 16, CN, CN, 0);
    actfinal_kernel<<<BN, 128, 0, stream>>>(actout, ss, W_s, b_s, out);
}
